// LSTNet_48292612276516
// MI455X (gfx1250) — matrix operand order repaired and checked by reading, not compiled, not run on hardware
//
#include <hip/hip_runtime.h>
#include <stdint.h>

// ---------------- types ----------------
typedef __attribute__((ext_vector_type(16))) __bf16  v16bf;
typedef __attribute__((ext_vector_type(8)))  float   v8f;
typedef __attribute__((ext_vector_type(4)))  unsigned int v4u;

union AF { v16bf v; v4u q[2]; };

#define MDIM 321
#define MIP  352
#define BATCH 256

__device__ __forceinline__ unsigned short f2bf(float f) {
  unsigned int u = __float_as_uint(f);
  u += 0x7FFFu + ((u >> 16) & 1u);
  return (unsigned short)(u >> 16);
}
__device__ __forceinline__ float sigm(float x) { return 1.f / (1.f + expf(-x)); }

// ---------------- staging: f32 (optionally strided/padded rows) -> bf16 padded cols ----------------
__global__ void k_stage(const float* __restrict__ src, unsigned short* __restrict__ dst,
                        int Tfull, int rowMul, int rowAdd, int Tsrc, int padL,
                        int Tout, int Msrc, int Mip_, long total)
{
  long idx = (long)blockIdx.x * 256 + threadIdx.x;
  if (idx >= total) return;
  int mi = (int)(idx % Mip_);
  long r  = idx / Mip_;
  int t  = (int)(r % Tout);
  int b  = (int)(r / Tout);
  int ts = t - padL; ts = ts < 0 ? 0 : (ts >= Tsrc ? Tsrc - 1 : ts);
  float v = 0.f;
  if (mi < Msrc)
    v = src[((long)b * Tfull + rowAdd + (long)ts * rowMul) * Msrc + mi];
  dst[idx] = f2bf(v);
}

// ---------------- generic K-tap conv as WMMA GEMM ----------------
// X bf16 (B, Tin, Mip), Wst bf16 [k][WMop][Mip] (zero padded), Y f32 (B, To, Mo)
// act: 0=leaky(0.01) 1=tanh 2=relu 3=none
__launch_bounds__(256)
__global__ void k_convgemm(const unsigned short* __restrict__ X,
                           const unsigned short* __restrict__ Wst,
                           const float* __restrict__ bias,
                           float* __restrict__ Y,
                           int Tin, int To, int Mip_, int Ktap, int WMop, int Mo, int act)
{
  __shared__ __align__(16) unsigned short As[264 * 32];
  __shared__ __align__(16) unsigned short Bs[6 * 32 * 32];

  const int b   = blockIdx.x;
  const int n0  = blockIdx.y * 32;
  const int tid = threadIdx.x;
  const int nM  = (To + 15) >> 4;               // <= 16
  const int AsRows = nM * 16 + Ktap - 1;        // <= 261
  const int lane = tid & 31, wv = tid >> 5;
  const int mrow = lane & 15, hb = lane >> 4;
  const int kbA = hb * 8, kbB = hb * 16;

  v8f acc[2][2];
  for (int s = 0; s < 2; ++s)
    for (int nh = 0; nh < 2; ++nh)
      for (int e = 0; e < 8; ++e) acc[s][nh][e] = 0.f;

  const unsigned short* Xb = X + (size_t)b * Tin * Mip_;
  const int nChunks = Mip_ >> 5;
  const int seg = tid & 3;

  for (int mc = 0; mc < nChunks; ++mc) {
    __syncthreads();
    // stage A rows (time window, per-batch contiguous; taps become LDS row offsets)
    for (int r = tid >> 2; r < AsRows; r += 64) {
      v4u v; v.x = 0u; v.y = 0u; v.z = 0u; v.w = 0u;
      if (r < Tin)
        v = *(const v4u*)(Xb + (size_t)r * Mip_ + mc * 32 + seg * 8);
      *(v4u*)&As[r * 32 + seg * 8] = v;
    }
    // stage B for all taps: Bs[k][n][mi] from Wst[k][n0+n][mi-chunk]
    const int nBRows = Ktap << 5;
    for (int r = tid >> 2; r < nBRows; r += 64) {
      int k = r >> 5, n = r & 31;
      v4u v = *(const v4u*)(Wst + (size_t)(k * WMop + n0 + n) * Mip_ + mc * 32 + seg * 8);
      *(v4u*)&Bs[r * 32 + seg * 8] = v;
    }
    __syncthreads();

    for (int s = 0; s < 2; ++s) {
      int msub = wv + s * 8;
      if (msub >= nM) continue;
      for (int k = 0; k < Ktap; ++k) {
        AF a;
        int ar = msub * 16 + mrow + k;
        a.q[0] = *(const v4u*)&As[ar * 32 + kbA];
        a.q[1] = *(const v4u*)&As[ar * 32 + kbA + 16];
        for (int nh = 0; nh < 2; ++nh) {
          AF bb;
          int br = (k * 32 + nh * 16 + mrow) * 32;
          bb.q[0] = *(const v4u*)&Bs[br + kbA];
          bb.q[1] = *(const v4u*)&Bs[br + kbA + 16];
          acc[s][nh] = __builtin_amdgcn_wmma_f32_16x16x32_bf16(
              false, a.v, false, bb.v, (short)0, acc[s][nh], false, false);
        }
      }
    }
  }

  // epilogue: bias + activation + guarded store
  for (int s = 0; s < 2; ++s) {
    int msub = wv + s * 8;
    if (msub >= nM) continue;
    for (int nh = 0; nh < 2; ++nh) {
      int col = n0 + nh * 16 + mrow;
      if (col >= Mo) continue;
      float bv = bias[col];
      for (int v = 0; v < 8; ++v) {
        int row = msub * 16 + hb * 8 + v;
        if (row < To) {
          float val = acc[s][nh][v] + bv;
          if      (act == 0) val = val > 0.f ? val : 0.01f * val;
          else if (act == 1) val = tanhf(val);
          else if (act == 2) val = fmaxf(val, 0.f);
          Y[((size_t)b * To + row) * Mo + col] = val;
        }
      }
    }
  }
}

// ---------------- elementwise ----------------
__global__ void k_dc(const float* __restrict__ z, int T, int Th,
                     const float* __restrict__ phi, const float* __restrict__ psi,
                     float* __restrict__ d, float* __restrict__ c, long total)
{
  long idx = (long)blockIdx.x * 256 + threadIdx.x;
  if (idx >= total) return;
  int m = (int)(idx % MDIM);
  long r = idx / MDIM;
  int t = (int)(r % Th);
  int b = (int)(r / Th);
  float e = z[((long)b * T + 2 * t) * MDIM + m];
  float o = z[((long)b * T + 2 * t + 1) * MDIM + m];
  d[idx] = o * expf(phi[idx]);
  c[idx] = e * expf(psi[idx]);
}

__global__ void k_euou(const float* __restrict__ c, const float* __restrict__ bU,
                       const float* __restrict__ d, const float* __restrict__ bP,
                       float* __restrict__ eu, float* __restrict__ ou, long total)
{
  long idx = (long)blockIdx.x * 256 + threadIdx.x;
  if (idx >= total) return;
  eu[idx] = c[idx] + bU[idx];
  ou[idx] = d[idx] - bP[idx];
}

__global__ void k_zip(const float* __restrict__ eu, const float* __restrict__ ou,
                      const float* __restrict__ xadd, float* __restrict__ out,
                      int L, long total)
{
  long idx = (long)blockIdx.x * 256 + threadIdx.x;
  if (idx >= total) return;
  int m = (int)(idx % MDIM);
  long r = idx / MDIM;
  int t2 = (int)(r % (2 * L));
  int b  = (int)(r / (2 * L));
  const float* s = (t2 & 1) ? ou : eu;
  float v = s[((long)b * L + (t2 >> 1)) * MDIM + m];
  if (xadd) v += xadd[idx];
  out[idx] = v;
}

__global__ void k_highway(const float* __restrict__ res, const float* __restrict__ xr,
                          const float* __restrict__ hww, const float* __restrict__ hwb,
                          float* __restrict__ out, long total)
{
  long idx = (long)blockIdx.x * 256 + threadIdx.x;
  if (idx >= total) return;
  int b = (int)(idx / MDIM), m = (int)(idx % MDIM);
  float a = hwb[0];
  for (int k = 0; k < 24; ++k)
    a += xr[((size_t)b * 192 + 168 + k) * MDIM + m] * hww[k];
  out[idx] = res[idx] + a;
}

// ---------------- weight conversion/staging ----------------
__global__ void k_cvt_tree(const float* __restrict__ src, unsigned short* __restrict__ dst,
                           int K, long total)
{
  long idx = (long)blockIdx.x * 256 + threadIdx.x;
  if (idx >= total) return;
  int mi = (int)(idx % 352);
  long r = idx / 352;
  int mo = (int)(r % 384); r /= 384;
  int k  = (int)(r % K);
  int ij = (int)(r / K);
  float v = 0.f;
  if (mo < 321 && mi < 321)
    v = src[(((long)ij * 321 + mo) * 321 + mi) * K + k];
  dst[idx] = f2bf(v);
}

__global__ void k_cvt_conv1(const float* __restrict__ src, unsigned short* __restrict__ dst, long total)
{
  long idx = (long)blockIdx.x * 256 + threadIdx.x;
  if (idx >= total) return;
  int mi = (int)(idx % 352);
  long r = idx / 352;
  int co = (int)(r % 128);
  int k  = (int)(r / 128);
  float v = (mi < 321) ? src[((long)co * 6 + k) * 321 + mi] : 0.f;
  dst[idx] = f2bf(v);
}

__global__ void k_cvt_mat(const float* __restrict__ src, unsigned short* __restrict__ dst,
                          int MoSrc, int Mi, long total)
{
  long idx = (long)blockIdx.x * 256 + threadIdx.x;
  if (idx >= total) return;
  int mi = (int)(idx % Mi);
  int mo = (int)(idx / Mi);
  float v = (mo < MoSrc) ? src[(long)mo * Mi + mi] : 0.f;
  dst[idx] = f2bf(v);
}

__global__ void k_transpose_whh(const float* __restrict__ src, float* __restrict__ dst, long total)
{
  long idx = (long)blockIdx.x * 256 + threadIdx.x;
  if (idx >= total) return;
  int g = (int)(idx % 384);
  int k = (int)(idx / 384);
  dst[idx] = src[(long)g * 128 + k];  // dst[k][g] = whh[g][k]
}

// ---------------- GRU1 hidden recurrence (gi precomputed via WMMA GEMM) ----------------
__launch_bounds__(128)
__global__ void k_gru1(const float* __restrict__ gi, const float* __restrict__ whhT,
                       const float* __restrict__ bhh, float* __restrict__ rout)
{
  int b = blockIdx.x, j = threadIdx.x;
  __shared__ float h[128];
  h[j] = 0.f; __syncthreads();
  for (int t = 0; t < 187; ++t) {
    float ar = bhh[j], az = bhh[128 + j], an = bhh[256 + j];
    for (int k = 0; k < 128; ++k) {
      float hk = h[k];
      const float* wr = whhT + (size_t)k * 384;
      ar += hk * wr[j];
      az += hk * wr[128 + j];
      an += hk * wr[256 + j];
    }
    const float* g = gi + ((size_t)b * 187 + t) * 384;
    float r  = sigm(g[j] + ar);
    float zt = sigm(g[128 + j] + az);
    float n  = tanhf(g[256 + j] + r * an);
    float hn = (1.f - zt) * n + zt * h[j];
    __syncthreads();
    h[j] = hn;
    __syncthreads();
  }
  rout[(size_t)b * 320 + j] = h[j];
}

// ---------------- skip GRU (hidden=8, 7 steps) ----------------
__global__ void k_grus(const float* __restrict__ cfeat,
                       const float* __restrict__ wih, const float* __restrict__ whh,
                       const float* __restrict__ bih, const float* __restrict__ bhh,
                       float* __restrict__ rout)
{
  int idx = blockIdx.x * 256 + threadIdx.x;
  if (idx >= 256 * 24) return;
  int b = idx / 24, sk = idx % 24;
  float h[8];
#pragma unroll
  for (int j = 0; j < 8; ++j) h[j] = 0.f;
  for (int p = 0; p < 7; ++p) {
    int t = 19 + p * 24 + sk;
    const float* x = cfeat + ((size_t)b * 187 + t) * 128;
    float gi[24];
#pragma unroll
    for (int g = 0; g < 24; ++g) {
      float a = bih[g];
      for (int k = 0; k < 128; ++k) a += x[k] * wih[g * 128 + k];
      gi[g] = a;
    }
    float hn[8];
#pragma unroll
    for (int j = 0; j < 8; ++j) {
      float hr = bhh[j], hz = bhh[8 + j], hnn = bhh[16 + j];
#pragma unroll
      for (int k = 0; k < 8; ++k) {
        hr  += h[k] * whh[j * 8 + k];
        hz  += h[k] * whh[(8 + j) * 8 + k];
        hnn += h[k] * whh[(16 + j) * 8 + k];
      }
      float r  = sigm(gi[j] + hr);
      float zt = sigm(gi[8 + j] + hz);
      float n  = tanhf(gi[16 + j] + r * hnn);
      hn[j] = (1.f - zt) * n + zt * h[j];
    }
#pragma unroll
    for (int j = 0; j < 8; ++j) h[j] = hn[j];
  }
#pragma unroll
  for (int j = 0; j < 8; ++j)
    rout[(size_t)b * 320 + 128 + sk * 8 + j] = h[j];
}

// ---------------- host ----------------
extern "C" void kernel_launch(void* const* d_in, const int* in_sizes, int n_in,
                              void* d_out, int out_size, void* d_ws, size_t ws_size,
                              hipStream_t stream)
{
  (void)in_sizes; (void)n_in; (void)out_size; (void)ws_size;
  const float* x      = (const float*)d_in[0];
  const float* tw1    = (const float*)d_in[1];
  const float* tb1    = (const float*)d_in[2];
  const float* tw2    = (const float*)d_in[3];
  const float* tb2    = (const float*)d_in[4];
  const float* c1w    = (const float*)d_in[5];
  const float* c1b    = (const float*)d_in[6];
  const float* g1_wih = (const float*)d_in[7];
  const float* g1_whh = (const float*)d_in[8];
  const float* g1_bih = (const float*)d_in[9];
  const float* g1_bhh = (const float*)d_in[10];
  const float* gs_wih = (const float*)d_in[11];
  const float* gs_whh = (const float*)d_in[12];
  const float* gs_bih = (const float*)d_in[13];
  const float* gs_bhh = (const float*)d_in[14];
  const float* l1w    = (const float*)d_in[15];
  const float* l1b    = (const float*)d_in[16];
  const float* hww    = (const float*)d_in[17];
  const float* hwb    = (const float*)d_in[18];
  float* out = (float*)d_out;

  char* ws = (char*)d_ws;
  size_t off = 0;
  auto carve = [&](size_t bytes) -> char* {
    char* p = ws + off;
    off = (off + bytes + 255) & ~(size_t)255;
    return p;
  };
  // f32 buffers
  float* xr    = (float*)carve((size_t)256 * 192 * 321 * 4);
  float* A96   = (float*)carve((size_t)256 * 96 * 321 * 4);
  float* B96b  = (float*)carve((size_t)256 * 96 * 321 * 4);
  float* C48   = (float*)carve((size_t)256 * 48 * 321 * 4);
  float* D48   = (float*)carve((size_t)256 * 48 * 321 * 4);
  float* E24   = (float*)carve((size_t)256 * 24 * 321 * 4);
  float* F24   = (float*)carve((size_t)256 * 24 * 321 * 4);
  float* phi   = (float*)carve((size_t)256 * 96 * 321 * 4);
  float* psi   = (float*)carve((size_t)256 * 96 * 321 * 4);
  float* dbuf  = (float*)carve((size_t)256 * 96 * 321 * 4);
  float* cbuf  = (float*)carve((size_t)256 * 96 * 321 * 4);
  float* t1    = (float*)carve((size_t)256 * 98 * 321 * 4);
  float* cfeat = (float*)carve((size_t)256 * 187 * 128 * 4);
  float* giall = (float*)carve((size_t)256 * 187 * 384 * 4);
  float* rcat  = (float*)carve((size_t)256 * 320 * 4);
  float* resb  = (float*)carve((size_t)256 * 321 * 4);
  float* whhT  = (float*)carve((size_t)128 * 384 * 4);
  // bf16 buffers
  unsigned short* sbuf  = (unsigned short*)carve((size_t)256 * 192 * 352 * 2);
  unsigned short* w1st  = (unsigned short*)carve((size_t)28 * 5 * 384 * 352 * 2);
  unsigned short* w2st  = (unsigned short*)carve((size_t)28 * 3 * 384 * 352 * 2);
  unsigned short* c1st  = (unsigned short*)carve((size_t)6 * 128 * 352 * 2);
  unsigned short* g1ist = (unsigned short*)carve((size_t)384 * 128 * 2);
  unsigned short* l1st  = (unsigned short*)carve((size_t)384 * 320 * 2);

  dim3 th(256);
  auto blocks1d = [](long total) { return dim3((unsigned int)((total + 255) / 256)); };

  auto stage = [&](const float* src, unsigned short* dst, int Bn, int Tfull, int rowMul,
                   int rowAdd, int Tsrc, int padL, int Tout, int Msrc, int Mip_) {
    long total = (long)Bn * Tout * Mip_;
    k_stage<<<blocks1d(total), th, 0, stream>>>(src, dst, Tfull, rowMul, rowAdd, Tsrc,
                                                padL, Tout, Msrc, Mip_, total);
  };

  // --- weight staging ---
  { long t = (long)28 * 5 * 384 * 352; k_cvt_tree<<<blocks1d(t), th, 0, stream>>>(tw1, w1st, 5, t); }
  { long t = (long)28 * 3 * 384 * 352; k_cvt_tree<<<blocks1d(t), th, 0, stream>>>(tw2, w2st, 3, t); }
  { long t = (long)6 * 128 * 352;      k_cvt_conv1<<<blocks1d(t), th, 0, stream>>>(c1w, c1st, t); }
  { long t = (long)384 * 128;          k_cvt_mat<<<blocks1d(t), th, 0, stream>>>(g1_wih, g1ist, 384, 128, t); }
  { long t = (long)384 * 320;          k_cvt_mat<<<blocks1d(t), th, 0, stream>>>(l1w, l1st, 321, 320, t); }
  { long t = (long)128 * 384;          k_transpose_whh<<<blocks1d(t), th, 0, stream>>>(g1_whh, whhT, t); }

  // --- SCINet tree ---
  auto run_branch = [&](const float* zsrc, int Tfull, int rowMul, int rowAdd, int Th,
                        int blk, int j, float* outBuf) {
    int Tin1 = Th + 6, To1 = Th + 2;
    stage(zsrc, sbuf, 256, Tfull, rowMul, rowAdd, Th, 3, Tin1, 321, 352);
    const unsigned short* w1p = w1st + (size_t)(blk * 4 + j) * 5 * 384 * 352;
    k_convgemm<<<dim3(256, 11), th, 0, stream>>>(sbuf, w1p, tb1 + (size_t)(blk * 4 + j) * 321,
                                                 t1, Tin1, To1, 352, 5, 384, 321, 0);
    stage(t1, sbuf, 256, To1, 1, 0, To1, 0, To1, 321, 352);
    const unsigned short* w2p = w2st + (size_t)(blk * 4 + j) * 3 * 384 * 352;
    k_convgemm<<<dim3(256, 11), th, 0, stream>>>(sbuf, w2p, tb2 + (size_t)(blk * 4 + j) * 321,
                                                 outBuf, To1, Th, 352, 3, 384, 321, 1);
  };

  auto run_block = [&](const float* z, int T, int blk, float* euO, float* ouO) {
    int Th = T / 2;
    long tot = (long)256 * Th * 321;
    run_branch(z, T, 2, 0, Th, blk, 0, phi);   // phi = branch(e)
    run_branch(z, T, 2, 1, Th, blk, 1, psi);   // psi = branch(o)
    k_dc<<<blocks1d(tot), th, 0, stream>>>(z, T, Th, phi, psi, dbuf, cbuf, tot);
    run_branch(dbuf, Th, 1, 0, Th, blk, 2, phi);  // bU -> phi
    run_branch(cbuf, Th, 1, 0, Th, blk, 3, psi);  // bP -> psi
    k_euou<<<blocks1d(tot), th, 0, stream>>>(cbuf, phi, dbuf, psi, euO, ouO, tot);
  };

  auto zip = [&](const float* eu, const float* ou, const float* xadd, float* o, int L) {
    long tot = (long)256 * 2 * L * 321;
    k_zip<<<blocks1d(tot), th, 0, stream>>>(eu, ou, xadd, o, L, tot);
  };

  run_block(x, 192, 0, A96, B96b);
  run_block(A96, 96, 1, C48, D48);
  run_block(C48, 48, 2, E24, F24);  zip(E24, F24, nullptr, C48, 24);
  run_block(D48, 48, 3, E24, F24);  zip(E24, F24, nullptr, D48, 24);
  zip(C48, D48, nullptr, A96, 48);
  run_block(B96b, 96, 4, C48, D48);
  run_block(C48, 48, 5, E24, F24);  zip(E24, F24, nullptr, C48, 24);
  run_block(D48, 48, 6, E24, F24);  zip(E24, F24, nullptr, D48, 24);
  zip(C48, D48, nullptr, B96b, 48);
  zip(A96, B96b, x, xr, 96);        // xr = x + res1

  // --- conv1 (6-tap, full-width) ---
  stage(xr, sbuf, 256, 192, 1, 0, 192, 0, 192, 321, 352);
  k_convgemm<<<dim3(256, 4), th, 0, stream>>>(sbuf, c1st, c1b, cfeat, 192, 187, 352, 6, 128, 128, 2);

  // --- GRU1: gi = c @ wih^T + bih via WMMA, then recurrence ---
  stage(cfeat, sbuf, 256, 187, 1, 0, 187, 0, 187, 128, 128);
  k_convgemm<<<dim3(256, 12), th, 0, stream>>>(sbuf, g1ist, g1_bih, giall, 187, 187, 128, 1, 384, 384, 3);
  k_gru1<<<dim3(256), dim3(128), 0, stream>>>(giall, whhT, g1_bhh, rcat);

  // --- skip GRU ---
  k_grus<<<blocks1d(6144), th, 0, stream>>>(cfeat, gs_wih, gs_whh, gs_bih, gs_bhh, rcat);

  // --- lin1 ---
  stage(rcat, sbuf, 1, 256, 1, 0, 256, 0, 256, 320, 320);
  k_convgemm<<<dim3(1, 11), th, 0, stream>>>(sbuf, l1st, l1b, resb, 256, 256, 320, 1, 384, 321, 3);

  // --- highway + output ---
  { long t = (long)256 * 321;
    k_highway<<<blocks1d(t), th, 0, stream>>>(resb, xr, hww, hwb, out, t); }
}
